// BasicVSR_LFN_32787780338197
// MI455X (gfx1250) — matrix operand order repaired and checked by reading, not compiled, not run on hardware
//
#include <hip/hip_runtime.h>
#include <hip/hip_bf16.h>

typedef __attribute__((ext_vector_type(16))) _Float16 v16h;
typedef __attribute__((ext_vector_type(8)))  _Float16 v8h;
typedef __attribute__((ext_vector_type(8)))  float    v8f;

#define B_   8
#define C_   96
#define H_   96
#define W_   160
#define HW_  (H_ * W_)
#define HH_  48
#define WH_  80
#define NPIX (B_ * HW_)          // 122880
#define NTIL (NPIX / 16)         // 7680 16-pixel tiles (W%16==0 -> tile on one row)

__device__ __forceinline__ float lrelu_f(float v) { return v >= 0.f ? v : 0.1f * v; }

// ---------------------------------------------------------------------------
// 1) Upsample flow: grouped ConvTranspose2d(2,2,k=4,s=2,p=1,groups=2) gather form
//    out[y][x] = sum_{ky,kx} dil_in[y-2+ky][x-2+kx] * w[c][0][3-ky][3-kx]
// ---------------------------------------------------------------------------
__global__ void k_upflow(const float* __restrict__ fp, const float* __restrict__ w,
                         float* __restrict__ flow) {
  int i = blockIdx.x * blockDim.x + threadIdx.x;
  if (i >= B_ * 2 * HW_) return;
  int x = i % W_, y = (i / W_) % H_, c = (i / HW_) % 2, b = i / (2 * HW_);
  float acc = 0.f;
  #pragma unroll
  for (int ky = 0; ky < 4; ++ky) {
    int du = y - 2 + ky;
    if (du < 0 || du > 2 * HH_ - 2 || (du & 1)) continue;
    #pragma unroll
    for (int kx = 0; kx < 4; ++kx) {
      int dv = x - 2 + kx;
      if (dv < 0 || dv > 2 * WH_ - 2 || (dv & 1)) continue;
      acc += fp[((b * 2 + c) * HH_ + (du >> 1)) * WH_ + (dv >> 1)]
           * w[c * 16 + (3 - ky) * 4 + (3 - kx)];
    }
  }
  flow[i] = acc;
}

// ---------------------------------------------------------------------------
// 2) feat_one f32 planar -> fp16 channels-last [b*HW + p][96]
// ---------------------------------------------------------------------------
__global__ void k_feat1_cl(const float* __restrict__ in, _Float16* __restrict__ out) {
  int i = blockIdx.x * blockDim.x + threadIdx.x;
  if (i >= NPIX * C_) return;
  int c = i % C_;
  int p = (i / C_) % HW_;
  int b = i / (C_ * HW_);
  out[i] = (_Float16)in[((size_t)(b * C_ + c)) * HW_ + p];
}

// ---------------------------------------------------------------------------
// 3) Backwarp feat_two by 2.5*flow (bilinear, zero pad). px = x + fx*W/(W-1).
//    Output fp16 channels-last [b*HW + p][96].
// ---------------------------------------------------------------------------
__global__ void k_warp(const float* __restrict__ feat2, const float* __restrict__ flow,
                       _Float16* __restrict__ wout) {
  int i = blockIdx.x * blockDim.x + threadIdx.x;
  if (i >= NPIX) return;
  int p = i % HW_, b = i / HW_;
  int x = p % W_, y = p / W_;
  float fx = flow[(b * 2 + 0) * HW_ + p] * 2.5f;
  float fy = flow[(b * 2 + 1) * HW_ + p] * 2.5f;
  float px = (float)x + fx * ((float)W_ / (float)(W_ - 1));
  float py = (float)y + fy * ((float)H_ / (float)(H_ - 1));
  float x0f = floorf(px), y0f = floorf(py);
  float wx1 = px - x0f, wx0 = 1.f - wx1;
  float wy1 = py - y0f, wy0 = 1.f - wy1;
  int x0 = (int)x0f, y0 = (int)y0f, x1 = x0 + 1, y1 = y0 + 1;
  bool vx0 = (x0 >= 0) & (x0 < W_), vx1 = (x1 >= 0) & (x1 < W_);
  bool vy0 = (y0 >= 0) & (y0 < H_), vy1 = (y1 >= 0) & (y1 < H_);
  int cx0 = min(max(x0, 0), W_ - 1), cx1 = min(max(x1, 0), W_ - 1);
  int cy0 = min(max(y0, 0), H_ - 1), cy1 = min(max(y1, 0), H_ - 1);
  float w00 = (vy0 && vx0) ? wy0 * wx0 : 0.f;
  float w01 = (vy0 && vx1) ? wy0 * wx1 : 0.f;
  float w10 = (vy1 && vx0) ? wy1 * wx0 : 0.f;
  float w11 = (vy1 && vx1) ? wy1 * wx1 : 0.f;
  int i00 = cy0 * W_ + cx0, i01 = cy0 * W_ + cx1;
  int i10 = cy1 * W_ + cx0, i11 = cy1 * W_ + cx1;
  const float* fb = feat2 + (size_t)b * C_ * HW_;
  _Float16* orow = wout + (size_t)i * C_;
  for (int c = 0; c < C_; ++c) {
    const float* fpl = fb + (size_t)c * HW_;
    float v = w00 * fpl[i00] + w01 * fpl[i01] + w10 * fpl[i10] + w11 * fpl[i11];
    orow[c] = (_Float16)v;
  }
}

// ---------------------------------------------------------------------------
// 4) Correlation (7x7 displacement, mean over 96 ch) + leaky relu.
//    Output fp16 channels-last, padded 49 -> 64 channels with zeros.
// ---------------------------------------------------------------------------
__global__ void k_corr(const _Float16* __restrict__ f1, const _Float16* __restrict__ wp,
                       _Float16* __restrict__ corr) {
  int i = blockIdx.x * blockDim.x + threadIdx.x;
  if (i >= NPIX) return;
  int p = i % HW_, b = i / HW_;
  int x = p % W_, y = p / W_;
  float a[C_];
  const v8h* ap = (const v8h*)(f1 + (size_t)i * C_);
  #pragma unroll
  for (int cc = 0; cc < C_ / 8; ++cc) {
    v8h t = ap[cc];
    #pragma unroll
    for (int j = 0; j < 8; ++j) a[cc * 8 + j] = (float)t[j];
  }
  _Float16* orow = corr + (size_t)i * 64;
  for (int o = 0; o < 49; ++o) {
    int dy = o / 7 - 3, dx = o % 7 - 3;
    int yy = y + dy, xx = x + dx;
    float acc = 0.f;
    if (yy >= 0 && yy < H_ && xx >= 0 && xx < W_) {
      const v8h* bp = (const v8h*)(wp + ((size_t)(b * HW_) + yy * W_ + xx) * C_);
      #pragma unroll
      for (int cc = 0; cc < C_ / 8; ++cc) {
        v8h t = bp[cc];
        #pragma unroll
        for (int j = 0; j < 8; ++j) acc += a[cc * 8 + j] * (float)t[j];
      }
    }
    orow[o] = (_Float16)lrelu_f(acc * (1.f / (float)C_));
  }
  #pragma unroll
  for (int o = 49; o < 64; ++o) orow[o] = (_Float16)0.f;
}

// ---------------------------------------------------------------------------
// 5) Weight reorder OIHW f32 -> [co][tap*CinPad + ci] fp16 (ci >= CinReal -> 0)
// ---------------------------------------------------------------------------
template <int CINR, int CINP, int COUT>
__global__ void k_wcvt(const float* __restrict__ w, _Float16* __restrict__ o) {
  int i = blockIdx.x * blockDim.x + threadIdx.x;
  if (i >= COUT * 9 * CINP) return;
  int ci = i % CINP;
  int t  = (i / CINP) % 9;
  int co = i / (CINP * 9);
  float v = (ci < CINR) ? w[(co * CINR + ci) * 9 + t] : 0.f;
  o[(size_t)co * (9 * CINP) + t * CINP + ci] = (_Float16)v;
}

// ---------------------------------------------------------------------------
// 6) Implicit-GEMM 3x3 conv (pad=1) on V_WMMA_F32_16X16X32_F16.
//    Wave tile: M=16 pixels (one row, since W%16==0) x N=16 cout, K=9*CIN.
//    A fragment (16-bit 16x32 layout): lane m=lane&15, hi=lane>=16.
//      lo lanes hold K {0..7,16..23}; hi lanes {8..15,24..31} -> per chunk all
//      from ONE pixel, two 16B channels-last loads (+hi*8 / +16+hi*8).
//    B fragment (32x16): lane n=lane&15, same K split as A (+hi*8 / +16+hi*8).
// ---------------------------------------------------------------------------
template <int CIN, int COUT, bool LRELU>
__global__ __launch_bounds__(128) void k_conv3x3(
    const _Float16* __restrict__ in,    // [NPIX][CIN] channels-last
    const _Float16* __restrict__ wt,    // [COUT][9*CIN]
    const float* __restrict__ bias,     // [COUT]
    _Float16* __restrict__ out,         // [NPIX][COUT] channels-last
    int nTiles) {
  constexpr int K = 9 * CIN;
  const int wave = threadIdx.x >> 5;
  const int lane = threadIdx.x & 31;
  const int hi   = lane >> 4;
  const int lm   = lane & 15;
  const int mt   = blockIdx.x * 4 + wave;
  if (mt >= nTiles) return;                 // wave-uniform
  const int g0 = mt * 16;                   // global pixel base = b*HW + p0
  const int b  = g0 / HW_;
  const int p0 = g0 - b * HW_;
  const int y  = p0 / W_;
  const int x0 = p0 - y * W_;               // multiple of 16 -> tile on one row
  const int xm = x0 + lm;                   // this lane's pixel column (A side)
  const int co = blockIdx.y * 16 + lm;      // this lane's cout (B/C side)
  const _Float16* wrow = wt + (size_t)co * K;
  v8f acc = {};
  #pragma unroll
  for (int tap = 0; tap < 9; ++tap) {
    const int ty = tap / 3 - 1, tx = tap % 3 - 1;
    const int yy = y + ty, xx = xm + tx;
    const bool valid = (yy >= 0) & (yy < H_) & (xx >= 0) & (xx < W_);
    for (int cc = 0; cc < CIN; cc += 32) {
      v16h afrag = {};
      if (valid) {
        const _Float16* arow = in + ((size_t)(b * HW_) + yy * W_ + xx) * CIN;
        const v8h* apx = (const v8h*)(arow + cc + hi * 8);
        ((v8h*)&afrag)[0] = apx[0];   // K 0..7   (lo) / 8..15  (hi)
        ((v8h*)&afrag)[1] = apx[2];   // K 16..23 (lo) / 24..31 (hi)
      }
      v16h bfrag;
      const v8h* bpx = (const v8h*)(wrow + tap * CIN + cc + hi * 8);
      ((v8h*)&bfrag)[0] = bpx[0];     // K 0..7 (lo) / 8..15 (hi)
      ((v8h*)&bfrag)[1] = bpx[2];
      acc = __builtin_amdgcn_wmma_f32_16x16x32_f16(
          false, afrag, false, bfrag, (short)0, acc, false, false);
    }
  }
  // C/D layout: VGPR r -> M = r + hi*8, N = lane&15
  const float bv = bias[co];
  #pragma unroll
  for (int r = 0; r < 8; ++r) {
    int mm = r + hi * 8;
    float v = acc[r] + bv;
    if (LRELU) v = lrelu_f(v);
    out[(size_t)(g0 + mm) * COUT + co] = (_Float16)v;
  }
}

// ---------------------------------------------------------------------------
// 7) Final 5x5 conv (32 -> 2, pad=2) + bias + flow add, f32 output (NCHW)
// ---------------------------------------------------------------------------
__global__ void k_conv5_out(const _Float16* __restrict__ h3,   // [NPIX][32]
                            const float* __restrict__ w4,      // [2][32][5][5]
                            const float* __restrict__ b4,
                            const float* __restrict__ flow,    // [B][2][HW]
                            float* __restrict__ out) {
  int i = blockIdx.x * blockDim.x + threadIdx.x;
  if (i >= NPIX) return;
  int p = i % HW_, b = i / HW_;
  int x = p % W_, y = p / W_;
  float a0 = b4[0], a1 = b4[1];
  for (int ky = 0; ky < 5; ++ky) {
    int yy = y + ky - 2;
    if (yy < 0 || yy >= H_) continue;
    for (int kx = 0; kx < 5; ++kx) {
      int xx = x + kx - 2;
      if (xx < 0 || xx >= W_) continue;
      int tap = ky * 5 + kx;
      const v8h* hp = (const v8h*)(h3 + ((size_t)(b * HW_) + yy * W_ + xx) * 32);
      #pragma unroll
      for (int cc = 0; cc < 4; ++cc) {
        v8h t = hp[cc];
        #pragma unroll
        for (int j = 0; j < 8; ++j) {
          int ci = cc * 8 + j;
          float hv = (float)t[j];
          a0 += hv * w4[(0 * 32 + ci) * 25 + tap];
          a1 += hv * w4[(1 * 32 + ci) * 25 + tap];
        }
      }
    }
  }
  out[(size_t)(b * 2 + 0) * HW_ + p] = flow[(b * 2 + 0) * HW_ + p] + a0;
  out[(size_t)(b * 2 + 1) * HW_ + p] = flow[(b * 2 + 1) * HW_ + p] + a1;
}

// ---------------------------------------------------------------------------
// Launch
// ---------------------------------------------------------------------------
extern "C" void kernel_launch(void* const* d_in, const int* in_sizes, int n_in,
                              void* d_out, int out_size, void* d_ws, size_t ws_size,
                              hipStream_t stream) {
  const float* feat_one  = (const float*)d_in[0];
  const float* feat_two  = (const float*)d_in[1];
  const float* flow_prev = (const float*)d_in[2];
  const float* up_w      = (const float*)d_in[3];
  const float* w1 = (const float*)d_in[4];  const float* b1 = (const float*)d_in[5];
  const float* w2 = (const float*)d_in[6];  const float* b2 = (const float*)d_in[7];
  const float* w3 = (const float*)d_in[8];  const float* b3 = (const float*)d_in[9];
  const float* w4 = (const float*)d_in[10]; const float* b4 = (const float*)d_in[11];
  float* out = (float*)d_out;

  char* ws = (char*)d_ws;
  // All offsets 256B aligned. Total ~114 MB.
  float*    flowF  = (float*)(ws + 0);                    //  983,040 B  [B][2][HW] f32
  _Float16* f1cl   = (_Float16*)(ws + 983040);            // 23,592,960 B [NPIX][96]
  _Float16* warpcl = (_Float16*)(ws + 24576000);          // 23,592,960 B [NPIX][96]
  _Float16* corrcl = (_Float16*)(ws + 48168960);          // 15,728,640 B [NPIX][64]
  _Float16* h1cl   = (_Float16*)(ws + 63897600);          // 31,457,280 B [NPIX][128]
  _Float16* h2cl   = (_Float16*)(ws + 95354880);          // 15,728,640 B [NPIX][64]
  _Float16* h3cl   = (_Float16*)(ws + 111083520);         //  7,864,320 B [NPIX][32]
  _Float16* w1h    = (_Float16*)(ws + 118947840);         //    147,456 B [128][576]
  _Float16* w2h    = (_Float16*)(ws + 119095296);         //    147,456 B [64][1152]
  _Float16* w3h    = (_Float16*)(ws + 119242752);         //     36,864 B [32][576]

  // 1. flow upsample
  k_upflow<<<(B_ * 2 * HW_ + 255) / 256, 256, 0, stream>>>(flow_prev, up_w, flowF);
  // 2. feat_one -> fp16 channels-last
  k_feat1_cl<<<(NPIX * C_ + 255) / 256, 256, 0, stream>>>(feat_one, f1cl);
  // 3. backwarp feat_two
  k_warp<<<(NPIX + 255) / 256, 256, 0, stream>>>(feat_two, flowF, warpcl);
  // 4. correlation + lrelu (padded to 64 ch)
  k_corr<<<(NPIX + 255) / 256, 256, 0, stream>>>(f1cl, warpcl, corrcl);
  // 5. weight reorders
  k_wcvt<49, 64, 128><<<(128 * 9 * 64 + 255) / 256, 256, 0, stream>>>(w1, w1h);
  k_wcvt<128, 128, 64><<<(64 * 9 * 128 + 255) / 256, 256, 0, stream>>>(w2, w2h);
  k_wcvt<64, 64, 32><<<(32 * 9 * 64 + 255) / 256, 256, 0, stream>>>(w3, w3h);
  // 6. WMMA convs: grid.x = tiles/4 waves per block, grid.y = cout/16
  dim3 blk(128, 1, 1);
  dim3 g1(NTIL / 4, 128 / 16, 1);
  k_conv3x3<64, 128, true><<<g1, blk, 0, stream>>>(corrcl, w1h, b1, h1cl, NTIL);
  dim3 g2(NTIL / 4, 64 / 16, 1);
  k_conv3x3<128, 64, true><<<g2, blk, 0, stream>>>(h1cl, w2h, b2, h2cl, NTIL);
  dim3 g3(NTIL / 4, 32 / 16, 1);
  k_conv3x3<64, 32, true><<<g3, blk, 0, stream>>>(h2cl, w3h, b3, h3cl, NTIL);
  // 7. 5x5 conv + flow add -> output
  k_conv5_out<<<(NPIX + 255) / 256, 256, 0, stream>>>(h3cl, w4, b4, flowF, out);

  (void)in_sizes; (void)n_in; (void)out_size; (void)ws_size;
}
